// BiLevelGAT_DSSE_82403242541628
// MI455X (gfx1250) — matrix operand order repaired and checked by reading, not compiled, not run on hardware
//
#include <hip/hip_runtime.h>
#include <hip/hip_bf16.h>

typedef __attribute__((ext_vector_type(16))) _Float16 v16h;
typedef __attribute__((ext_vector_type(8)))  _Float16 v8h;
typedef __attribute__((ext_vector_type(8)))  float    v8f;

#define WGD 96
#define EDIMK 8

static __device__ __forceinline__ unsigned fkey(float f) {
    unsigned u = __float_as_uint(f);
    return (u & 0x80000000u) ? ~u : (u | 0x80000000u);
}
static __device__ __forceinline__ float keyf(unsigned u) {
    u = (u & 0x80000000u) ? (u & 0x7fffffffu) : ~u;
    return __uint_as_float(u);
}

// ---------------------------------------------------------------------------
// WMMA GEMM: C[M,Ncols] = act(A[M,K] @ B[K,Ncols] + bias)
// A row-major f16 (row stride = K); BT is B TRANSPOSED, [Ncols, K] row-major,
// so both A and B fragments load as contiguous 16-byte b128 chunks.
// One wave -> one 16x16 tile; K is compile-time -> fully unrolled WMMA chain.
// ---------------------------------------------------------------------------
template <int K>
__global__ void gemm_wmma_f16_kernel(const _Float16* __restrict__ A,
                                     const _Float16* __restrict__ BT,
                                     const float* __restrict__ bias,
                                     float* __restrict__ C,
                                     int M, int Ncols, int act_mode)
{
    const int lane   = threadIdx.x & 31;
    const int wave   = threadIdx.x >> 5;
    const int ntiles = Ncols >> 4;
    const int mtiles = (M + 15) >> 4;
    const int t = blockIdx.x * (blockDim.x >> 5) + wave;
    if (t >= mtiles * ntiles) return;          // uniform per wave
    const int mtile = t / ntiles;
    const int ntile = t % ntiles;
    const int kg = lane >> 4;                  // half-wave id

    // A fragment (16-bit A 16x32 layout): lane row = lane&15;
    // halves K = k0 + kg*8 + [0..7]  and  K = k0 + 16 + kg*8 + [0..7]  (contiguous)
    int arow = mtile * 16 + (lane & 15);
    if (arow >= M) arow = M - 1;               // clamp keeps EXEC full for WMMA
    const _Float16* Arow = A + (size_t)arow * K;

    // B fragment: lane col = lane&15; halves K = k0 + kg*8 + [0..7]  and  K = k0 + 16 + kg*8 + [0..7] (contiguous in BT)
    const int bcol = ntile * 16 + (lane & 15);
    const _Float16* Bcol = BT + (size_t)bcol * K;

    v8f acc = {};
#pragma unroll
    for (int k0 = 0; k0 < K; k0 += 32) {
        const v8h alo = *(const v8h*)(Arow + k0 + kg * 8);
        const v8h ahi = *(const v8h*)(Arow + k0 + 16 + kg * 8);
        const v8h blo = *(const v8h*)(Bcol + k0 + kg * 8);
        const v8h bhi = *(const v8h*)(Bcol + k0 + 16 + kg * 8);
        const v16h a = __builtin_shufflevector(alo, ahi, 0,1,2,3,4,5,6,7,8,9,10,11,12,13,14,15);
        const v16h b = __builtin_shufflevector(blo, bhi, 0,1,2,3,4,5,6,7,8,9,10,11,12,13,14,15);
        acc = __builtin_amdgcn_wmma_f32_16x16x32_f16(
            /*neg_a=*/false, a, /*neg_b=*/false, b,
            /*c_mod=*/(short)0, acc, /*reuse_a=*/false, /*reuse_b=*/false);
    }

    const float bb = bias ? bias[bcol] : 0.0f;
    float* Crow = C + (size_t)(mtile * 16 + kg * 8) * Ncols + bcol;
    const bool edgeTile = ((M & 15) != 0) && (mtile == mtiles - 1);  // uniform
    if (!edgeTile) {
#pragma unroll
        for (int r = 0; r < 8; ++r) {
            float v = acc[r] + bb;
            if (act_mode == 1) v = (v > 0.0f) ? v : 0.01f * v;
            Crow[(size_t)r * Ncols] = v;
        }
    } else {
#pragma unroll
        for (int r = 0; r < 8; ++r) {
            const int row = mtile * 16 + kg * 8 + r;
            if (row < M) {
                float v = acc[r] + bb;
                if (act_mode == 1) v = (v > 0.0f) ? v : 0.01f * v;
                Crow[(size_t)r * Ncols] = v;
            }
        }
    }
}

// ---------------------------------------------------------------------------
// Elementwise helpers
// ---------------------------------------------------------------------------
__global__ void zero_u32_kernel(unsigned* __restrict__ p, int n) {
    int i = blockIdx.x * blockDim.x + threadIdx.x;
    if (i < n) p[i] = 0u;
}

__global__ void f32_to_f16_kernel(const float* __restrict__ in,
                                  _Float16* __restrict__ out, int n) {
    int i = blockIdx.x * blockDim.x + threadIdx.x;
    if (i < n) out[i] = (_Float16)in[i];
}

// weight convert + transpose: in [Kd, Nd] f32 -> out [Nd, Kd] f16
__global__ void f32_to_f16_T_kernel(const float* __restrict__ in,
                                    _Float16* __restrict__ out,
                                    int Kd, int Nd) {
    int i = blockIdx.x * blockDim.x + threadIdx.x;
    if (i >= Kd * Nd) return;
    const int k = i / Nd, n = i % Nd;
    out[(size_t)n * Kd + k] = (_Float16)in[i];
}

// act (optional leaky 0.01) + f32 -> f16 with output stride (for concat packing)
__global__ void act_to_f16_kernel(const float* __restrict__ in,
                                  _Float16* __restrict__ out,
                                  int n, int width, int outStride, int act_mode) {
    int i = blockIdx.x * blockDim.x + threadIdx.x;
    if (i >= n) return;
    const int r = i / width, c = i % width;
    float v = in[i];
    if (act_mode == 1) v = (v > 0.0f) ? v : 0.01f * v;
    out[(size_t)r * outStride + c] = (_Float16)v;
}

__global__ void init_bias_kernel(float* __restrict__ out,
                                 const float* __restrict__ b, int n, int width) {
    int i = blockIdx.x * blockDim.x + threadIdx.x;
    if (i < n) out[i] = b[i % width];
}

// mean-accumulator for edge_attr (sums; consumers multiply by 1/E)
__global__ void edge_mean_kernel(const float* __restrict__ ea,
                                 float* __restrict__ sums, int E) {
    int t = blockIdx.x * blockDim.x + threadIdx.x;
    int stride = gridDim.x * blockDim.x;
    float local[EDIMK];
#pragma unroll
    for (int j = 0; j < EDIMK; ++j) local[j] = 0.0f;
    for (int e = t; e < E; e += stride)
#pragma unroll
        for (int j = 0; j < EDIMK; ++j) local[j] += ea[(size_t)e * EDIMK + j];
#pragma unroll
    for (int j = 0; j < EDIMK; ++j) atomicAdd(sums + j, local[j]);
}

// ---------------------------------------------------------------------------
// Edge phase: one wave32 per edge; each lane covers dims {lane, lane+32, lane+64}
// ---------------------------------------------------------------------------
__global__ void edge_logit_kernel(const float* __restrict__ hl,
                                  const float* __restrict__ hr,
                                  const int* __restrict__ srcA,
                                  const int* __restrict__ dstA,
                                  const float* __restrict__ ea,
                                  const float* __restrict__ meansum, float invE,
                                  const float* __restrict__ We,
                                  const float* __restrict__ att,
                                  float* __restrict__ logit,
                                  unsigned* __restrict__ maxkey,
                                  int E_real, int Etot)
{
    const int wid  = (blockIdx.x * blockDim.x + threadIdx.x) >> 5;
    const int lane = threadIdx.x & 31;
    if (wid >= Etot) return;
    int s, d;
    float eav[EDIMK];
    if (wid < E_real) {
        s = srcA[wid]; d = dstA[wid];
#pragma unroll
        for (int j = 0; j < EDIMK; ++j) eav[j] = ea[(size_t)wid * EDIMK + j];
    } else {
        s = d = wid - E_real;                 // self loop, mean edge_attr
#pragma unroll
        for (int j = 0; j < EDIMK; ++j) eav[j] = meansum[j] * invE;
    }
    float part = 0.0f;
#pragma unroll
    for (int c = 0; c < 3; ++c) {
        const int dim = lane + c * 32;
        float m = hl[(size_t)s * WGD + dim] + hr[(size_t)d * WGD + dim];
#pragma unroll
        for (int j = 0; j < EDIMK; ++j) m += eav[j] * We[j * WGD + dim];
        const float e = (m > 0.0f) ? m : 0.2f * m;   // GATv2 slope
        part += e * att[dim];
    }
#pragma unroll
    for (int off = 16; off > 0; off >>= 1)
        part += __shfl_down(part, off, 32);
    if (lane == 0) {
        logit[wid] = part;
        atomicMax(maxkey + d, fkey(part));
    }
}

__global__ void edge_expsum_kernel(const float* __restrict__ logit,
                                   const int* __restrict__ dstA,
                                   const unsigned* __restrict__ maxkey,
                                   float* __restrict__ ex,
                                   float* __restrict__ den,
                                   int E_real, int Etot)
{
    int e = blockIdx.x * blockDim.x + threadIdx.x;
    if (e >= Etot) return;
    const int d = (e < E_real) ? dstA[e] : (e - E_real);
    const float v = __expf(logit[e] - keyf(maxkey[d]));
    ex[e] = v;
    atomicAdd(den + d, v);
}

__global__ void edge_scatter_kernel(const float* __restrict__ hl,
                                    const int* __restrict__ srcA,
                                    const int* __restrict__ dstA,
                                    const float* __restrict__ ex,
                                    const float* __restrict__ den,
                                    float* __restrict__ out,
                                    int E_real, int Etot)
{
    const int wid  = (blockIdx.x * blockDim.x + threadIdx.x) >> 5;
    const int lane = threadIdx.x & 31;
    if (wid >= Etot) return;
    int s, d;
    if (wid < E_real) { s = srcA[wid]; d = dstA[wid]; }
    else             { s = d = wid - E_real; }
    const float alpha = ex[wid] / den[d];
#pragma unroll
    for (int c = 0; c < 3; ++c) {
        const int dim = lane + c * 32;
        atomicAdd(out + (size_t)d * WGD + dim, alpha * hl[(size_t)s * WGD + dim]);
    }
}

// final 256 -> 2 head (tiny; VALU)
__global__ void pred2_kernel(const float* __restrict__ hid,
                             const float* __restrict__ W2,
                             const float* __restrict__ b2,
                             float* __restrict__ out, int Nn)
{
    int n = blockIdx.x * blockDim.x + threadIdx.x;
    if (n >= Nn) return;
    float o0 = b2[0], o1 = b2[1];
    for (int k = 0; k < 256; ++k) {
        const float h = hid[(size_t)n * 256 + k];
        o0 += h * W2[k * 2 + 0];
        o1 += h * W2[k * 2 + 1];
    }
    out[(size_t)n * 2 + 0] = o0;
    out[(size_t)n * 2 + 1] = o1;
}

// ---------------------------------------------------------------------------
// Host driver
// ---------------------------------------------------------------------------
static inline int cdiv(long long a, long long b) { return (int)((a + b - 1) / b); }

extern "C" void kernel_launch(void* const* d_in, const int* in_sizes, int n_in,
                              void* d_out, int out_size, void* d_ws, size_t ws_size,
                              hipStream_t stream)
{
    const float* x_f32   = (const float*)d_in[0];
    const int*   eidx    = (const int*)d_in[1];
    const float* ea      = (const float*)d_in[2];
    const float* lWl     = (const float*)d_in[3];
    const float* lWr     = (const float*)d_in[4];
    const float* lWe     = (const float*)d_in[5];
    const float* lAtt    = (const float*)d_in[6];
    const float* lB      = (const float*)d_in[7];
    const float* gWl     = (const float*)d_in[8];
    const float* gWr     = (const float*)d_in[9];
    const float* gWe     = (const float*)d_in[10];
    const float* gAtt    = (const float*)d_in[11];
    const float* gB      = (const float*)d_in[12];
    const float* fusW    = (const float*)d_in[13];
    const float* fusB    = (const float*)d_in[14];
    const float* pW1     = (const float*)d_in[15];
    const float* pB1     = (const float*)d_in[16];
    const float* pW2     = (const float*)d_in[17];
    const float* pB2     = (const float*)d_in[18];
    float* out = (float*)d_out;

    const int N    = in_sizes[0] / WGD;
    const int E    = in_sizes[1] / 2;
    const int Etot = E + N;
    const int Lnum = 3;
    const int* srcA = eidx;        // edge_index row 0
    const int* dstA = eidx + E;    // edge_index row 1

    // ---- workspace layout (bytes, 256-aligned) ----
    char* ws = (char*)d_ws;
    size_t off = 0;
    auto alloc = [&](size_t bytes) -> char* {
        char* p = ws + off;
        off = (off + bytes + 255) & ~(size_t)255;
        return p;
    };
    float*     meansum = (float*)alloc(EDIMK * sizeof(float));
    _Float16*  xf16    = (_Float16*)alloc((size_t)N * WGD * 2);
    _Float16*  cur16   = (_Float16*)alloc((size_t)N * WGD * 2);
    _Float16*  Wl16[2]; Wl16[0] = (_Float16*)alloc((size_t)Lnum * WGD * WGD * 2);
                        Wl16[1] = (_Float16*)alloc((size_t)Lnum * WGD * WGD * 2);
    _Float16*  Wr16[2]; Wr16[0] = (_Float16*)alloc((size_t)Lnum * WGD * WGD * 2);
                        Wr16[1] = (_Float16*)alloc((size_t)Lnum * WGD * WGD * 2);
    _Float16*  fusW16  = (_Float16*)alloc((size_t)2 * WGD * WGD * 2);
    _Float16*  pW116   = (_Float16*)alloc((size_t)WGD * 256 * 2);
    float*     hlbuf   = (float*)alloc((size_t)N * WGD * 4);
    float*     hrbuf   = (float*)alloc((size_t)N * WGD * 4);
    float*     accum   = (float*)alloc((size_t)N * WGD * 4);
    float*     logit   = (float*)alloc((size_t)Etot * 4);
    float*     exbuf   = (float*)alloc((size_t)Etot * 4);
    unsigned*  maxkey  = (unsigned*)alloc((size_t)N * 4);
    float*     den     = (float*)alloc((size_t)N * 4);
    _Float16*  cat16   = (_Float16*)alloc((size_t)N * 2 * WGD * 2);
    float*     hid     = (float*)alloc((size_t)N * 256 * 4);
    (void)ws_size;

    const int BS = 256;

    // ---- per-call conversions (weights go in TRANSPOSED for b128 fragment loads) ----
    zero_u32_kernel<<<1, 32, 0, stream>>>((unsigned*)meansum, EDIMK);
    edge_mean_kernel<<<256, BS, 0, stream>>>(ea, meansum, E);

    f32_to_f16_kernel<<<cdiv((size_t)N * WGD, BS), BS, 0, stream>>>(x_f32, xf16, N * WGD);

    const int wmat = WGD * WGD;                 // one 96x96 layer matrix
    const int wgrid = cdiv(wmat, BS);
    for (int l = 0; l < Lnum; ++l) {
        f32_to_f16_T_kernel<<<wgrid, BS, 0, stream>>>(lWl + (size_t)l * wmat, Wl16[0] + (size_t)l * wmat, WGD, WGD);
        f32_to_f16_T_kernel<<<wgrid, BS, 0, stream>>>(gWl + (size_t)l * wmat, Wl16[1] + (size_t)l * wmat, WGD, WGD);
        f32_to_f16_T_kernel<<<wgrid, BS, 0, stream>>>(lWr + (size_t)l * wmat, Wr16[0] + (size_t)l * wmat, WGD, WGD);
        f32_to_f16_T_kernel<<<wgrid, BS, 0, stream>>>(gWr + (size_t)l * wmat, Wr16[1] + (size_t)l * wmat, WGD, WGD);
    }
    f32_to_f16_T_kernel<<<cdiv(2 * WGD * WGD, BS), BS, 0, stream>>>(fusW, fusW16, 2 * WGD, WGD);
    f32_to_f16_T_kernel<<<cdiv(WGD * 256, BS), BS, 0, stream>>>(pW1, pW116, WGD, 256);

    auto gemm = [&](const _Float16* A, const _Float16* BT, const float* bias,
                    float* C, int M, int Nc, int K, int act) {
        const long long tiles = (long long)((M + 15) / 16) * (Nc / 16);
        const int grid = cdiv(tiles, 8);
        if (K == 96)
            gemm_wmma_f16_kernel<96><<<grid, BS, 0, stream>>>(A, BT, bias, C, M, Nc, act);
        else
            gemm_wmma_f16_kernel<192><<<grid, BS, 0, stream>>>(A, BT, bias, C, M, Nc, act);
    };

    const int edgeWaveBlocks = cdiv((long long)Etot * 32, BS);
    const float invE = 1.0f / (float)E;

    // ---- two GATv2 branches ----
    for (int branch = 0; branch < 2; ++branch) {
        const _Float16* Wl = Wl16[branch];
        const _Float16* Wr = Wr16[branch];
        const float* We  = branch ? gWe  : lWe;
        const float* att = branch ? gAtt : lAtt;
        const float* bia = branch ? gB   : lB;

        for (int l = 0; l < Lnum; ++l) {
            const _Float16* hin = (l == 0) ? xf16 : cur16;

            gemm(hin, Wl + (size_t)l * wmat, nullptr, hlbuf, N, WGD, WGD, 0);
            gemm(hin, Wr + (size_t)l * wmat, nullptr, hrbuf, N, WGD, WGD, 0);

            zero_u32_kernel<<<cdiv(N, BS), BS, 0, stream>>>(maxkey, N);
            zero_u32_kernel<<<cdiv(N, BS), BS, 0, stream>>>((unsigned*)den, N);
            init_bias_kernel<<<cdiv((size_t)N * WGD, BS), BS, 0, stream>>>(
                accum, bia + (size_t)l * WGD, N * WGD, WGD);

            edge_logit_kernel<<<edgeWaveBlocks, BS, 0, stream>>>(
                hlbuf, hrbuf, srcA, dstA, ea, meansum, invE,
                We + (size_t)l * EDIMK * WGD, att + (size_t)l * WGD,
                logit, maxkey, E, Etot);
            edge_expsum_kernel<<<cdiv(Etot, BS), BS, 0, stream>>>(
                logit, dstA, maxkey, exbuf, den, E, Etot);
            edge_scatter_kernel<<<edgeWaveBlocks, BS, 0, stream>>>(
                hlbuf, srcA, dstA, exbuf, den, accum, E, Etot);

            if (l < Lnum - 1) {
                act_to_f16_kernel<<<cdiv((size_t)N * WGD, BS), BS, 0, stream>>>(
                    accum, cur16, N * WGD, WGD, WGD, 1);
            } else {
                // final layer output -> its half of the concat buffer [N, 192]
                act_to_f16_kernel<<<cdiv((size_t)N * WGD, BS), BS, 0, stream>>>(
                    accum, cat16 + branch * WGD, N * WGD, WGD, 2 * WGD, 1);
            }
        }
    }

    // ---- fusion + prediction head ----
    gemm(cat16, fusW16, fusB, hlbuf /*fused f32*/, N, WGD, 2 * WGD, 1);
    act_to_f16_kernel<<<cdiv((size_t)N * WGD, BS), BS, 0, stream>>>(
        hlbuf, cur16 /*fused f16*/, N * WGD, WGD, WGD, 0);
    gemm(cur16, pW116, pB1, hid, N, 256, WGD, 1);
    pred2_kernel<<<cdiv(N, BS), BS, 0, stream>>>(hid, pW2, pB2, out, N);
}
